// DiffAttention_90572270338250
// MI455X (gfx1250) — compile-verified, matrix operand order checked by reading, not run on hardware
//
#include <hip/hip_runtime.h>
#include <hip/hip_bf16.h>

typedef __bf16 bf16_t;
typedef __attribute__((ext_vector_type(16))) __bf16 v16bf;
typedef __attribute__((ext_vector_type(8)))  __bf16 v8bf;
typedef __attribute__((ext_vector_type(8)))  float  v8f;

#define LAMBDA_INIT 0.7008206670670481f
#define RMS_EPS 1e-5f

// ---------------------------------------------------------------------------
// Async global->LDS 16-byte copy (per-lane), tracked by ASYNCcnt.
// LDS address = low 32 bits of the generic pointer (ISA: generic LDS address
// is {SHARED_BASE, offset32}); global address is a 64-bit VGPR pair.
// ---------------------------------------------------------------------------
__device__ __forceinline__ void async_ld16(void* lds_ptr, const void* gptr) {
    unsigned lo = (unsigned)(unsigned long long)lds_ptr;
    unsigned long long ga = (unsigned long long)gptr;
    asm volatile("global_load_async_to_lds_b128 %0, %1, off"
                 :: "v"(lo), "v"(ga) : "memory");
}
__device__ __forceinline__ void wait_async0() {
    asm volatile("s_wait_asynccnt 0x0" ::: "memory");
}

// ---------------------------------------------------------------------------
// WMMA fragment loader (bf16, 16x32 A-operand / 32x16 B-operand).
// Per CDNA5 ISA 7.12.2: lane L owns row (L&15); its 16 K-elements live in two
// contiguous 8-element (16B) chunks at K = (L>>4)*8 and K = 16 + (L>>4)*8.
// ---------------------------------------------------------------------------
__device__ __forceinline__ v16bf load_frag(const bf16_t* rowbase, int stride, int k0) {
    const int lane = threadIdx.x & 31;
    const int r  = lane & 15;
    const int kh = lane >> 4;
    const bf16_t* rp = rowbase + r * stride + k0 + kh * 8;
    v8bf c0 = *(const v8bf*)(rp);
    v8bf c1 = *(const v8bf*)(rp + 16);
    v16bf f;
#pragma unroll
    for (int i = 0; i < 8; ++i) { f[i] = c0[i]; f[i + 8] = c1[i]; }
    return f;
}

__device__ __forceinline__ v8f wmma_bf16(v16bf a, v16bf b, v8f c) {
    return __builtin_amdgcn_wmma_f32_16x16x32_bf16(false, a, false, b, (short)0, c,
                                                   false, false);
}

// ---------------------------------------------------------------------------
// fp32 -> bf16 conversion (vector x4)
// ---------------------------------------------------------------------------
__global__ __launch_bounds__(256) void cvt_bf16_kernel(const float* __restrict__ in,
                                                       bf16_t* __restrict__ out, int n4) {
    int i = blockIdx.x * blockDim.x + threadIdx.x;
    if (i < n4) {
        float4 v = ((const float4*)in)[i];
        bf16_t o0 = (bf16_t)v.x, o1 = (bf16_t)v.y, o2 = (bf16_t)v.z, o3 = (bf16_t)v.w;
        bf16_t* p = out + i * 4;
        p[0] = o0; p[1] = o1; p[2] = o2; p[3] = o3;
    }
}

// ---------------------------------------------------------------------------
// lambda = exp(sum lq1*lk1) - exp(sum lq2*lk2) + LAMBDA_INIT   (hd = 64)
// ---------------------------------------------------------------------------
__global__ void lam_kernel(const float* __restrict__ lq1, const float* __restrict__ lk1,
                           const float* __restrict__ lq2, const float* __restrict__ lk2,
                           float* __restrict__ lam_out) {
    int t = threadIdx.x;  // 32 threads
    float s1 = lq1[t] * lk1[t] + lq1[t + 32] * lk1[t + 32];
    float s2 = lq2[t] * lk2[t] + lq2[t + 32] * lk2[t + 32];
#pragma unroll
    for (int off = 16; off; off >>= 1) {
        s1 += __shfl_xor(s1, off);
        s2 += __shfl_xor(s2, off);
    }
    if (t == 0) *lam_out = __expf(s1) - __expf(s2) + LAMBDA_INIT;
}

// ---------------------------------------------------------------------------
// GEMM: C[M,N] = A[M,K] @ W[N,K]^T   (bf16 in, WMMA f32 accum)
// 128x128 tile, BK=32, double-buffered async global->LDS staging.
// 8 waves in 2(M)x4(N) grid, each wave 64x32 (4x2 WMMA tiles).
// mode 0: fp32 row-major out;  mode 1: bf16 (B,S,16,64)->(B,16,S,64) scatter;
// mode 2: bf16 (B,S,8,128)->(B,8,S,128) scatter.   scale applied pre-store.
// ---------------------------------------------------------------------------
__global__ __launch_bounds__(256) void gemm_wmma_kernel(
    const bf16_t* __restrict__ A, const bf16_t* __restrict__ W,
    void* __restrict__ Cout, int M, int N, int K, int mode, float scale) {
    __shared__ __align__(16) bf16_t As[2][128 * 40];
    __shared__ __align__(16) bf16_t Bs[2][128 * 40];

    const int tid  = threadIdx.x;
    const int lane = tid & 31;
    const int kh   = lane >> 4, ln = lane & 15;
    const int w    = tid >> 5;
    const int wm   = w & 1, wn = w >> 1;
    const int bn   = blockIdx.x, bm = blockIdx.y;
    const long arow0 = (long)bm * 128;
    const long brow0 = (long)bn * 128;

    // staging: 512 chunks of 8 bf16 per matrix; this thread owns chunks 2t,2t+1
    const int c0 = tid * 2,      row0 = c0 >> 2, cc0 = (c0 & 3) << 3;
    const int c1 = tid * 2 + 1,  row1 = c1 >> 2, cc1 = (c1 & 3) << 3;
    const bf16_t* gA0 = &A[(arow0 + row0) * K + cc0];
    const bf16_t* gA1 = &A[(arow0 + row1) * K + cc1];
    const bf16_t* gB0 = &W[(brow0 + row0) * K + cc0];
    const bf16_t* gB1 = &W[(brow0 + row1) * K + cc1];

    auto stage = [&](int buf, int kt) {
        async_ld16(&As[buf][row0 * 40 + cc0], gA0 + kt);
        async_ld16(&As[buf][row1 * 40 + cc1], gA1 + kt);
        async_ld16(&Bs[buf][row0 * 40 + cc0], gB0 + kt);
        async_ld16(&Bs[buf][row1 * 40 + cc1], gB1 + kt);
    };

    v8f acc[4][2];
#pragma unroll
    for (int i = 0; i < 4; ++i)
#pragma unroll
        for (int j = 0; j < 2; ++j)
#pragma unroll
            for (int r = 0; r < 8; ++r) acc[i][j][r] = 0.0f;

    stage(0, 0);
    for (int kt = 0; kt < K; kt += 32) {
        const int buf = (kt >> 5) & 1;
        wait_async0();        // stage(buf) complete (issued last iteration)
        __syncthreads();      // all waves' stage(buf) visible; prior reads done
        if (kt + 32 < K) stage(buf ^ 1, kt + 32);  // overlaps with WMMAs below

        v16bf af[4], bfr[2];
#pragma unroll
        for (int nt = 0; nt < 2; ++nt)
            bfr[nt] = load_frag(&Bs[buf][(wn * 32 + nt * 16) * 40], 40, 0);
#pragma unroll
        for (int mt = 0; mt < 4; ++mt)
            af[mt] = load_frag(&As[buf][(wm * 64 + mt * 16) * 40], 40, 0);
#pragma unroll
        for (int mt = 0; mt < 4; ++mt)
#pragma unroll
            for (int nt = 0; nt < 2; ++nt)
                acc[mt][nt] = wmma_bf16(af[mt], bfr[nt], acc[mt][nt]);
        __syncthreads();      // readers of buf done before buf is re-staged
    }

#pragma unroll
    for (int mt = 0; mt < 4; ++mt)
#pragma unroll
        for (int nt = 0; nt < 2; ++nt)
#pragma unroll
            for (int r = 0; r < 8; ++r) {
                int mg = bm * 128 + wm * 64 + mt * 16 + r + 8 * kh;
                int ng = bn * 128 + wn * 32 + nt * 16 + ln;
                float v = acc[mt][nt][r] * scale;
                if (mode == 0) {
                    ((float*)Cout)[(long)mg * N + ng] = v;
                } else if (mode == 1) {       // q/k: (b, j=n/64, s, d=n%64)
                    int b = mg >> 11, s = mg & 2047;
                    int j = ng >> 6,  d = ng & 63;
                    ((bf16_t*)Cout)[((((long)b * 16 + j) << 11) + s) * 64 + d] = (bf16_t)v;
                } else {                      // v: (b, h=n/128, s, d=n%128)
                    int b = mg >> 11, s = mg & 2047;
                    int h = ng >> 7,  d = ng & 127;
                    ((bf16_t*)Cout)[((((long)b * 8 + h) << 11) + s) * 128 + d] = (bf16_t)v;
                }
            }
}

// ---------------------------------------------------------------------------
// Fused differential flash attention.
// Grid: B*H*(S/64) blocks of 256 threads (8 waves).
// wave w: component c = w>>2 (sub-head 2h+c), q-subtile qs = w&3 (16 rows).
// K staged async to LDS (both components), V staged transposed, P routed via
// wave-private LDS.  Epilogue: O0 - lam*O1, RMSNorm(128), *(1-LAMBDA_INIT).
// ---------------------------------------------------------------------------
__global__ __launch_bounds__(256) void diffattn_kernel(
    const bf16_t* __restrict__ qb, const bf16_t* __restrict__ kb,
    const bf16_t* __restrict__ vb, bf16_t* __restrict__ ob,
    const float* __restrict__ lam_p) {
    __shared__ __align__(16) char smem[55296];
    bf16_t* Kt  = (bf16_t*)smem;              // [2 comps * 64 keys][72] of d
    bf16_t* Vt  = (bf16_t*)(smem + 18432);    // [128 dims][72] of keys (V^T)
    bf16_t* Pb  = (bf16_t*)(smem + 36864);    // [8 waves][16][72]
    float*  Cmb = (float*)smem;               // [64][128] reused after the loop

    const int tid  = threadIdx.x;
    const int lane = tid & 31;
    const int kh   = lane >> 4, ln = lane & 15;
    const int w    = tid >> 5;
    const int c    = w >> 2, qs = w & 3;

    int bx = blockIdx.x;
    const int qblk = bx & 31; bx >>= 5;
    const int h = bx & 7;
    const int b = bx >> 3;
    const int q0 = qblk * 64;
    const float lam = *lam_p;

    const int j = 2 * h + c;
    const bf16_t* qbase = qb + ((((long)b * 16 + j) << 11) + q0 + qs * 16) * 64;
    const bf16_t* vbase = vb + (((long)b * 8 + h) << 11) * 128;

    // Q fragments for this wave's 16 rows (k-steps 0 and 32 of hd=64)
    v16bf qf0 = load_frag(qbase, 64, 0);
    v16bf qf1 = load_frag(qbase, 64, 32);

    // K async-staging addresses: 1024 chunks of 8 bf16 (2 comps x 64 keys x 8)
    // thread owns chunks 4t..4t+3; chunk -> row = comp*64+key (8 chunks/row)
    int krow[4], kdc[4];
    const bf16_t* kgp[4];
#pragma unroll
    for (int i = 0; i < 4; ++i) {
        int cid = tid * 4 + i;
        krow[i] = cid >> 3;
        kdc[i]  = (cid & 7) << 3;
        int comp = krow[i] >> 6, key = krow[i] & 63;
        kgp[i] = kb + ((((long)b * 16 + 2 * h + comp) << 11) + key) * 64 + kdc[i];
    }

    float mrow[8], lrow[8];
    v8f o[8];
#pragma unroll
    for (int r = 0; r < 8; ++r) { mrow[r] = -1e30f; lrow[r] = 0.0f; }
#pragma unroll
    for (int nt = 0; nt < 8; ++nt)
#pragma unroll
        for (int r = 0; r < 8; ++r) o[nt][r] = 0.0f;

    const int nkb = qblk + 1;
    for (int kbk = 0; kbk < nkb; ++kbk) {
        __syncthreads();   // prior iteration's readers of Kt/Vt are done
        // async-stage K tile (both components) for this key block
#pragma unroll
        for (int i = 0; i < 4; ++i)
            async_ld16(&Kt[krow[i] * 72 + kdc[i]], kgp[i] + (long)kbk * 64 * 64);
        // stage V^T: Vt[dim][key], 64x128 tile, 256 threads x 4 chunks of 8
#pragma unroll
        for (int i = 0; i < 4; ++i) {
            int cid = tid * 4 + i;
            int key = cid >> 4, dc = (cid & 15) << 3;
            v8bf vv = *(const v8bf*)&vbase[(long)(kbk * 64 + key) * 128 + dc];
#pragma unroll
            for (int e = 0; e < 8; ++e) Vt[(dc + e) * 72 + key] = vv[e];
        }
        wait_async0();
        __syncthreads();

        // scores: S[16 x 64] = Q(16x64) @ K(64keys x 64)^T
        v8f s[4];
#pragma unroll
        for (int nt = 0; nt < 4; ++nt)
#pragma unroll
            for (int r = 0; r < 8; ++r) s[nt][r] = 0.0f;

#pragma unroll
        for (int nt = 0; nt < 4; ++nt) {
            v16bf kf0 = load_frag(Kt + (c * 64 + nt * 16) * 72, 72, 0);
            v16bf kf1 = load_frag(Kt + (c * 64 + nt * 16) * 72, 72, 32);
            s[nt] = wmma_bf16(qf0, kf0, s[nt]);
            s[nt] = wmma_bf16(qf1, kf1, s[nt]);
        }

        if (kbk == qblk) {  // causal mask on the diagonal tile
#pragma unroll
            for (int nt = 0; nt < 4; ++nt)
#pragma unroll
                for (int r = 0; r < 8; ++r) {
                    int mg = qs * 16 + r + 8 * kh;
                    int ng = nt * 16 + ln;
                    if (ng > mg) s[nt][r] = -1e30f;
                }
        }

        // online softmax (row stats replicated over each 16-lane half)
        float esc[8];
#pragma unroll
        for (int r = 0; r < 8; ++r) {
            float mx = fmaxf(fmaxf(s[0][r], s[1][r]), fmaxf(s[2][r], s[3][r]));
            mx = fmaxf(mx, __shfl_xor(mx, 1));
            mx = fmaxf(mx, __shfl_xor(mx, 2));
            mx = fmaxf(mx, __shfl_xor(mx, 4));
            mx = fmaxf(mx, __shfl_xor(mx, 8));
            float mnew = fmaxf(mrow[r], mx);
            esc[r] = __expf(mrow[r] - mnew);
            mrow[r] = mnew;
#pragma unroll
            for (int nt = 0; nt < 4; ++nt) s[nt][r] = __expf(s[nt][r] - mnew);
            float rs = s[0][r] + s[1][r] + s[2][r] + s[3][r];
            rs += __shfl_xor(rs, 1);
            rs += __shfl_xor(rs, 2);
            rs += __shfl_xor(rs, 4);
            rs += __shfl_xor(rs, 8);
            lrow[r] = lrow[r] * esc[r] + rs;
        }
#pragma unroll
        for (int nt = 0; nt < 8; ++nt)
#pragma unroll
            for (int r = 0; r < 8; ++r) o[nt][r] *= esc[r];

        // spill P (bf16) to wave-private LDS, reload as A fragments
        bf16_t* pw = Pb + w * (16 * 72);
#pragma unroll
        for (int nt = 0; nt < 4; ++nt)
#pragma unroll
            for (int r = 0; r < 8; ++r)
                pw[(r + 8 * kh) * 72 + nt * 16 + ln] = (bf16_t)s[nt][r];

        v16bf pf0 = load_frag(pw, 72, 0);
        v16bf pf1 = load_frag(pw, 72, 32);
#pragma unroll
        for (int nt = 0; nt < 8; ++nt) {
            v16bf vf0 = load_frag(Vt + nt * 16 * 72, 72, 0);
            v16bf vf1 = load_frag(Vt + nt * 16 * 72, 72, 32);
            o[nt] = wmma_bf16(pf0, vf0, o[nt]);
            o[nt] = wmma_bf16(pf1, vf1, o[nt]);
        }
    }

    // ---- combine components, RMSNorm, write ----
    __syncthreads();
    if (c == 1) {
#pragma unroll
        for (int nt = 0; nt < 8; ++nt)
#pragma unroll
            for (int r = 0; r < 8; ++r)
                Cmb[(qs * 16 + r + 8 * kh) * 128 + nt * 16 + ln] = o[nt][r] / lrow[r];
    }
    __syncthreads();
    if (c == 0) {
        float ssq[8];
#pragma unroll
        for (int r = 0; r < 8; ++r) ssq[r] = 0.0f;
#pragma unroll
        for (int nt = 0; nt < 8; ++nt)
#pragma unroll
            for (int r = 0; r < 8; ++r) {
                float v = o[nt][r] / lrow[r] -
                          lam * Cmb[(qs * 16 + r + 8 * kh) * 128 + nt * 16 + ln];
                o[nt][r] = v;
                ssq[r] += v * v;
            }
#pragma unroll
        for (int r = 0; r < 8; ++r) {
            float t = ssq[r];
            t += __shfl_xor(t, 1);
            t += __shfl_xor(t, 2);
            t += __shfl_xor(t, 4);
            t += __shfl_xor(t, 8);
            ssq[r] = rsqrtf(t * (1.0f / 128.0f) + RMS_EPS) * (1.0f - LAMBDA_INIT);
        }
#pragma unroll
        for (int nt = 0; nt < 8; ++nt)
#pragma unroll
            for (int r = 0; r < 8; ++r) {
                int sg = q0 + qs * 16 + r + 8 * kh;
                ob[((long)(b * 2048 + sg)) * 1024 + h * 128 + nt * 16 + ln] =
                    (bf16_t)(o[nt][r] * ssq[r]);
            }
    }
}

// ---------------------------------------------------------------------------
extern "C" void kernel_launch(void* const* d_in, const int* in_sizes, int n_in,
                              void* d_out, int out_size, void* d_ws, size_t ws_size,
                              hipStream_t stream) {
    const float* x   = (const float*)d_in[0];
    const float* Wq  = (const float*)d_in[1];
    const float* Wk  = (const float*)d_in[2];
    const float* Wv  = (const float*)d_in[3];
    const float* Wo  = (const float*)d_in[4];
    const float* lq1 = (const float*)d_in[5];
    const float* lk1 = (const float*)d_in[6];
    const float* lq2 = (const float*)d_in[7];
    const float* lk2 = (const float*)d_in[8];

    const int M = 4096, D = 1024;  // B*S, DIM
    char* ws = (char*)d_ws;
    size_t off = 0;
    bf16_t* xb  = (bf16_t*)(ws + off); off += (size_t)M * D * 2;
    bf16_t* Wqb = (bf16_t*)(ws + off); off += (size_t)D * D * 2;
    bf16_t* Wkb = (bf16_t*)(ws + off); off += (size_t)D * D * 2;
    bf16_t* Wvb = (bf16_t*)(ws + off); off += (size_t)D * D * 2;
    bf16_t* Wob = (bf16_t*)(ws + off); off += (size_t)D * D * 2;
    bf16_t* qbf = (bf16_t*)(ws + off); off += (size_t)M * D * 2;
    bf16_t* kbf = (bf16_t*)(ws + off); off += (size_t)M * D * 2;
    bf16_t* vbf = (bf16_t*)(ws + off); off += (size_t)M * D * 2;
    bf16_t* obf = (bf16_t*)(ws + off); off += (size_t)M * D * 2;
    float*  lam = (float*)(ws + off);  off += 16;

    cvt_bf16_kernel<<<(M * D / 4 + 255) / 256, 256, 0, stream>>>(x,  xb,  M * D / 4);
    cvt_bf16_kernel<<<(D * D / 4 + 255) / 256, 256, 0, stream>>>(Wq, Wqb, D * D / 4);
    cvt_bf16_kernel<<<(D * D / 4 + 255) / 256, 256, 0, stream>>>(Wk, Wkb, D * D / 4);
    cvt_bf16_kernel<<<(D * D / 4 + 255) / 256, 256, 0, stream>>>(Wv, Wvb, D * D / 4);
    cvt_bf16_kernel<<<(D * D / 4 + 255) / 256, 256, 0, stream>>>(Wo, Wob, D * D / 4);
    lam_kernel<<<1, 32, 0, stream>>>(lq1, lk1, lq2, lk2, lam);

    dim3 gg(D / 128, M / 128);
    // q scaled by hd^-0.5 = 0.125 at projection time (matches reference)
    gemm_wmma_kernel<<<gg, 256, 0, stream>>>(xb, Wqb, qbf, M, D, D, 1, 0.125f);
    gemm_wmma_kernel<<<gg, 256, 0, stream>>>(xb, Wkb, kbf, M, D, D, 1, 1.0f);
    gemm_wmma_kernel<<<gg, 256, 0, stream>>>(xb, Wvb, vbf, M, D, D, 2, 1.0f);

    diffattn_kernel<<<2 * 8 * 32, 256, 0, stream>>>(qbf, kbf, vbf, obf, lam);

    gemm_wmma_kernel<<<gg, 256, 0, stream>>>(obf, Wob, d_out, M, D, D, 0, 1.0f);
}
